// LSTMBlock_65738769433282
// MI455X (gfx1250) — matrix operand order repaired and checked by reading, not compiled, not run on hardware
//
#include <hip/hip_runtime.h>
#include <math.h>

// ---------------- problem constants ----------------
#define Bsz 32
#define Tsz 512
#define Dsz 512
#define Hsz 1024
#define NWG 64          // persistent workgroups (8 waves each -> 512 waves = 512 GEMM tiles)

typedef __bf16 bf16_t;
typedef __attribute__((ext_vector_type(16))) __bf16 v16bf;
typedef __attribute__((ext_vector_type(8)))  __bf16 v8bf;
typedef __attribute__((ext_vector_type(8)))  float  v8f;
typedef __attribute__((ext_vector_type(4)))  unsigned int v4u;
typedef int v4i_t __attribute__((vector_size(16)));

#if defined(__has_builtin)
#if __has_builtin(__builtin_amdgcn_global_load_async_to_lds_b128)
#define HAVE_ASYNC_LDS 1
#endif
#endif

// ---------------- workspace layout (bytes) ----------------
// Wbf   : 4*H*H bf16          =  8 MB
// Ubf   : 4*H*D bf16          =  4 MB
// Xbf   : B*T*D bf16          = 16 MB
// xproj : 4*T*B*H bf16        = 128 MB   (bU folded in)
// preact: 4*B*H f32           = 512 KB
// h_bf  : B*H bf16            = 64 KB
// c_st  : B*H f32             = 128 KB
// counter: 256 B
static constexpr size_t OFF_WBF   = 0;
static constexpr size_t OFF_UBF   = OFF_WBF   + 4ull*Hsz*Hsz*2;
static constexpr size_t OFF_XBF   = OFF_UBF   + 4ull*Hsz*Dsz*2;
static constexpr size_t OFF_XPROJ = OFF_XBF   + (size_t)Bsz*Tsz*Dsz*2;
static constexpr size_t OFF_PRE   = OFF_XPROJ + 4ull*Tsz*Bsz*Hsz*2;
static constexpr size_t OFF_HBF   = OFF_PRE   + 4ull*Bsz*Hsz*4;
static constexpr size_t OFF_CST   = OFF_HBF   + (size_t)Bsz*Hsz*2;
static constexpr size_t OFF_CNT   = OFF_CST   + (size_t)Bsz*Hsz*4;

// ---------------- helpers ----------------
__global__ __launch_bounds__(256) void cvt_f32_to_bf16(const float* __restrict__ src,
                                                       bf16_t* __restrict__ dst, int n) {
    for (int i = blockIdx.x * 256 + threadIdx.x; i < n; i += gridDim.x * 256)
        dst[i] = (bf16_t)src[i];
}

// B-fragment (32x16, bf16): element (k,n) = p[n*ld + k]; lane holds one column n,
// K kb..kb+7 and kb+16..kb+23 (kb = 0 for lanes 0-15, 8 for lanes 16-31) -> two 16B loads.
__device__ inline v16bf load_b_frag(const bf16_t* __restrict__ p, int ld, int lane) {
    const int n  = lane & 15;
    const int kb = (lane < 16) ? 0 : 8;
    union { v16bf v; v8bf h[2]; } u;
    const bf16_t* q = p + (size_t)n * ld + kb;
    u.h[0] = *reinterpret_cast<const v8bf*>(q);
    u.h[1] = *reinterpret_cast<const v8bf*>(q + 16);
    return u.v;
}

// ---------------- input projection: xproj[g][t][b][h] = X[b,t,:]·U_g[h,:] + bU_g[h] ----------------
// Each wave computes a 16x64 output strip (4 N-tiles) so the A-fragment is reused 4x.
__global__ __launch_bounds__(256) void proj_kernel(const bf16_t* __restrict__ Xbf,
                                                   const bf16_t* __restrict__ Ubf,
                                                   const float* __restrict__ bU0,
                                                   const float* __restrict__ bU1,
                                                   const float* __restrict__ bU2,
                                                   const float* __restrict__ bU3,
                                                   bf16_t* __restrict__ xproj) {
    const int lane = threadIdx.x & 31;
    const int w    = (blockIdx.x * 256 + threadIdx.x) >> 5;   // global wave id
    const int g    = w / ((Bsz * Tsz / 16) * (Hsz / 64));
    const int rem  = w % ((Bsz * Tsz / 16) * (Hsz / 64));
    const int mt   = rem / (Hsz / 64);                        // 0..1023 (rows of [B*T])
    const int ntg  = rem % (Hsz / 64);                        // 0..15  (group of 4 n-tiles)

    const bf16_t* Arow = Xbf + (size_t)(mt * 16) * Dsz;       // 16 rows, ld = D
    const bf16_t* Bpan = Ubf + (size_t)g * Hsz * Dsz + (size_t)(ntg * 64) * Dsz; // ld = D

    const int arow = lane & 15;
    const int koff = (lane < 16) ? 0 : 8;

    v8f acc[4] = {{}, {}, {}, {}};
#pragma unroll 2
    for (int kk = 0; kk < Dsz / 32; ++kk) {
        union { v16bf v; v8bf h[2]; } ua;
        const bf16_t* ap = Arow + (size_t)arow * Dsz + kk * 32 + koff;
        ua.h[0] = *reinterpret_cast<const v8bf*>(ap);
        ua.h[1] = *reinterpret_cast<const v8bf*>(ap + 16);
#pragma unroll
        for (int j = 0; j < 4; ++j) {
            v16bf bf = load_b_frag(Bpan + (size_t)(j * 16) * Dsz + kk * 32, Dsz, lane);
            acc[j] = __builtin_amdgcn_wmma_f32_16x16x32_bf16(false, ua.v, false, bf,
                                                             (short)0, acc[j], false, false);
        }
    }

    const float* bU = (g == 0) ? bU0 : (g == 1) ? bU1 : (g == 2) ? bU2 : bU3;
    const int col = lane & 15;
    const int rb  = (lane < 16) ? 0 : 8;
#pragma unroll
    for (int j = 0; j < 4; ++j) {
        const int n  = (ntg * 4 + j) * 16 + col;
        const float bu = bU[n];
#pragma unroll
        for (int r = 0; r < 8; ++r) {
            const int grow = mt * 16 + rb + r;     // flattened (b,t): b-major
            const int b = grow / Tsz;
            const int t = grow % Tsz;
            xproj[(((size_t)g * Tsz + t) * Bsz + b) * Hsz + n] = (bf16_t)(acc[j][r] + bu);
        }
    }
}

// ---------------- grid barrier (monotonic counter, reset each launch) ----------------
__device__ inline void grid_sync(unsigned* counter, unsigned target) {
    __threadfence();
    __syncthreads();
    if (threadIdx.x == 0) {
        __hip_atomic_fetch_add(counter, 1u, __ATOMIC_ACQ_REL, __HIP_MEMORY_SCOPE_AGENT);
        while (__hip_atomic_load(counter, __ATOMIC_ACQUIRE, __HIP_MEMORY_SCOPE_AGENT) < target)
            __builtin_amdgcn_s_sleep(2);
    }
    __syncthreads();
}

// ---------------- persistent recurrent kernel ----------------
__global__ __launch_bounds__(256) void lstm_kernel(const bf16_t* __restrict__ Wbf,
                                                   const bf16_t* __restrict__ xproj,
                                                   const float* __restrict__ bWf,
                                                   const float* __restrict__ bWi,
                                                   const float* __restrict__ bWo,
                                                   const float* __restrict__ bWc,
                                                   float* __restrict__ preact,
                                                   bf16_t* __restrict__ h_bf,
                                                   float* __restrict__ c_st,
                                                   unsigned* __restrict__ counter,
                                                   float* __restrict__ out) {
    __shared__ __align__(32) bf16_t sh_h[Bsz * Hsz];          // 64 KB: full h state in LDS

    const int tid   = threadIdx.x;
    const int lane  = tid & 31;
    const int wave  = tid >> 5;
    const int wtask = blockIdx.x * 8 + wave;                  // 0..511
    const int g  = wtask >> 7;                                // gate
    const int mt = (wtask >> 6) & 1;                          // batch-row tile (0..1)
    const int nt = wtask & 63;                                // out-col tile

    const bf16_t* Wpanel = Wbf + (size_t)g * Hsz * Hsz + (size_t)(nt * 16) * Hsz; // ld = H
    const int arow = mt * 16 + (lane & 15);
    const int koff = (lane < 16) ? 0 : 8;
    const int nloc = lane & 15;
    const int gtid = blockIdx.x * 256 + tid;
    constexpr int BH = Bsz * Hsz;

    for (int t = 0; t < Tsz; ++t) {
        // stage h (bf16) into LDS: 4096 x 16B
#ifdef HAVE_ASYNC_LDS
        {
            v4i_t __attribute__((address_space(1)))* gp =
                (v4i_t __attribute__((address_space(1)))*)(bf16_t*)h_bf;
            v4i_t __attribute__((address_space(3)))* lp =
                (v4i_t __attribute__((address_space(3)))*)sh_h;
            for (int i = tid; i < BH / 8; i += 256)
                __builtin_amdgcn_global_load_async_to_lds_b128(gp + i, lp + i, 0, 0);
#if __has_builtin(__builtin_amdgcn_s_wait_asynccnt)
            __builtin_amdgcn_s_wait_asynccnt(0);
#else
            asm volatile("s_wait_asynccnt 0x0" ::: "memory");
#endif
        }
#else
        for (int i = tid; i < BH / 8; i += 256)
            reinterpret_cast<v4u*>(sh_h)[i] = reinterpret_cast<const v4u*>(h_bf)[i];
#endif
        __syncthreads();

        // GEMM: pre = h @ W_g^T  (M=32 split x2, N=1024 split x64, K=1024)
        v8f acc = {};
#pragma unroll 4
        for (int kk = 0; kk < Hsz / 32; ++kk) {
            // warm L0 for the W stream a few iterations ahead (global_prefetch_b8)
            __builtin_prefetch(Wpanel + (size_t)nloc * Hsz + (kk + 4) * 32, 0, 1);
            union { v16bf v; v8bf h[2]; } ua;
            const bf16_t* ap = &sh_h[arow * Hsz + kk * 32 + koff];
            ua.h[0] = *reinterpret_cast<const v8bf*>(ap);      // K koff..koff+7   (ds load)
            ua.h[1] = *reinterpret_cast<const v8bf*>(ap + 16); // K koff+16..+23
            v16bf bf = load_b_frag(Wpanel + kk * 32, Hsz, lane); // from L2-resident W
            acc = __builtin_amdgcn_wmma_f32_16x16x32_bf16(false, ua.v, false, bf,
                                                          (short)0, acc, false, false);
        }
        {
            const int col = lane & 15;
            const int rb  = (lane < 16) ? 0 : 8;
#pragma unroll
            for (int r = 0; r < 8; ++r) {
                const int brow = mt * 16 + rb + r;
                preact[(size_t)g * BH + (size_t)brow * Hsz + nt * 16 + col] = acc[r];
            }
        }
        grid_sync(counter, (unsigned)((2 * t + 1) * NWG));

        // elementwise gate math: 2 elements per thread over B*H = 32768
#pragma unroll
        for (int rep = 0; rep < 2; ++rep) {
            const int e  = gtid + rep * (NWG * 256);
            const int b  = e >> 10;           // / H
            const int hc = e & (Hsz - 1);
            const size_t xi = ((size_t)t * Bsz + b) * Hsz + hc;
            float pf = preact[0 * BH + e] + bWf[hc] + (float)xproj[0ull * Tsz * Bsz * Hsz + xi];
            float pi = preact[1 * BH + e] + bWi[hc] + (float)xproj[1ull * Tsz * Bsz * Hsz + xi];
            float po = preact[2 * BH + e] + bWo[hc] + (float)xproj[2ull * Tsz * Bsz * Hsz + xi];
            float pc = preact[3 * BH + e] + bWc[hc] + (float)xproj[3ull * Tsz * Bsz * Hsz + xi];
            const float fg = 1.0f / (1.0f + __expf(-pf));
            const float ig = 1.0f / (1.0f + __expf(-pi));
            const float og = 1.0f / (1.0f + __expf(-po));
            const float cb = tanhf(pc);
            const float cn = fg * c_st[e] + ig * cb;
            const float hn = og * tanhf(cn);
            c_st[e] = cn;
            h_bf[e] = (bf16_t)hn;
            out[(size_t)b * Tsz * Hsz + (size_t)t * Hsz + hc] = hn;
        }
        grid_sync(counter, (unsigned)((2 * t + 2) * NWG));
    }
}

// ---------------- host launcher ----------------
extern "C" void kernel_launch(void* const* d_in, const int* in_sizes, int n_in,
                              void* d_out, int out_size, void* d_ws, size_t ws_size,
                              hipStream_t stream) {
    const float* X = (const float*)d_in[0];
    const float* Wg[4] = { (const float*)d_in[1], (const float*)d_in[5],
                           (const float*)d_in[9], (const float*)d_in[13] };
    const float* bW[4] = { (const float*)d_in[2], (const float*)d_in[6],
                           (const float*)d_in[10], (const float*)d_in[14] };
    const float* Ug[4] = { (const float*)d_in[3], (const float*)d_in[7],
                           (const float*)d_in[11], (const float*)d_in[15] };
    const float* bU[4] = { (const float*)d_in[4], (const float*)d_in[8],
                           (const float*)d_in[12], (const float*)d_in[16] };

    char* ws = (char*)d_ws;
    bf16_t*   Wbf    = (bf16_t*)(ws + OFF_WBF);
    bf16_t*   Ubf    = (bf16_t*)(ws + OFF_UBF);
    bf16_t*   Xbf    = (bf16_t*)(ws + OFF_XBF);
    bf16_t*   xproj  = (bf16_t*)(ws + OFF_XPROJ);
    float*    preact = (float*) (ws + OFF_PRE);
    bf16_t*   h_bf   = (bf16_t*)(ws + OFF_HBF);
    float*    c_st   = (float*) (ws + OFF_CST);
    unsigned* cnt    = (unsigned*)(ws + OFF_CNT);

    (void)hipMemsetAsync(cnt, 0, 256, stream);
    (void)hipMemsetAsync(h_bf, 0, (size_t)Bsz * Hsz * 2, stream);
    (void)hipMemsetAsync(c_st, 0, (size_t)Bsz * Hsz * 4, stream);

    for (int g = 0; g < 4; ++g)
        cvt_f32_to_bf16<<<512, 256, 0, stream>>>(Wg[g], Wbf + (size_t)g * Hsz * Hsz, Hsz * Hsz);
    for (int g = 0; g < 4; ++g)
        cvt_f32_to_bf16<<<256, 256, 0, stream>>>(Ug[g], Ubf + (size_t)g * Hsz * Dsz, Hsz * Dsz);
    cvt_f32_to_bf16<<<2048, 256, 0, stream>>>(X, Xbf, Bsz * Tsz * Dsz);

    // 4 gates * 1024 row-tiles * 16 n-groups = 65536 waves -> 8192 blocks
    proj_kernel<<<8192, 256, 0, stream>>>(Xbf, Ubf, bU[0], bU[1], bU[2], bU[3], xproj);

    lstm_kernel<<<NWG, 256, 0, stream>>>(Wbf, xproj, bW[0], bW[1], bW[2], bW[3],
                                         preact, h_bf, c_st, cnt, (float*)d_out);
}
